// ASSAAttention_4148938408558
// MI455X (gfx1250) — matrix operand order repaired and checked by reading, not compiled, not run on hardware
//
#include <hip/hip_runtime.h>

// ASSA attention (sparse relu^2 + dense softmax, shared QK^T), flash style,
// bf16 WMMA f32-accumulate on gfx1250 (wave32).
// Fast path: bf16 prepass into workspace + Tensor Data Mover (TDM) tile DMA
// into LDS (TENSORcnt-synchronized). Fallback: legacy f32 staging kernel.

typedef __attribute__((ext_vector_type(16))) __bf16 v16bf;
typedef __attribute__((ext_vector_type(8)))  float  v8f;
typedef __attribute__((ext_vector_type(2)))  __bf16 bf16x2;
typedef __attribute__((ext_vector_type(2)))  float  f32x2;
typedef __attribute__((ext_vector_type(4)))  unsigned int u32x4;
typedef __attribute__((ext_vector_type(8)))  int    i32x8;
typedef __attribute__((ext_vector_type(4)))  int    i32x4;

union FragAB { v16bf v; unsigned int u[8]; };
union FragC  { v8f  v; float f[8]; };

// packed f32x2 -> bf16x2 (RNE): v_cvt_pk_bf16_f32
__device__ __forceinline__ unsigned int cvt2(float lo, float hi) {
    f32x2 f; f.x = lo; f.y = hi;
    bf16x2 h = __builtin_convertvector(f, bf16x2);
    return __builtin_bit_cast(unsigned int, h);
}

#define B_    2
#define L_    2048
#define S_    2048
#define H_    8
#define E_    64
#define NIT   (S_ / 32)
#define KSTR  72   // K-tile LDS row stride, bf16 (64 + 8 pad) == 128B + 16B pad
#define VSTR  40   // V^T-tile LDS row stride, bf16 (32 + 8 pad) == 64B + 16B pad
#define PSTRU 20   // P-buffer column stride, dwords (16 rows + 4 pad)

#define LOG2E 1.4426950408889634f
#define LN2SQ 0.4804530139182014f  // ln(2)^2

// TDM availability (guard also protects the host compilation pass)
#if defined(__has_builtin)
#  if __has_builtin(__builtin_amdgcn_tensor_load_to_lds)
#    define HAVE_TDM 1
#  endif
#endif
#ifndef HAVE_TDM
#  define HAVE_TDM 0
#endif

// A-matrix (16-bit, 16x32) pair-slot -> K mapping (contiguous, matches B fills):
// lanes 0-15: {0,2,...,14}; lanes 16-31: +16
__device__ __forceinline__ int a_pair_k(int p, int half) {
    return 2 * p + half * 16;
}

#if HAVE_TDM
// Issue one 2D TDM tile load: bf16 (data_size=1 -> 2B), LDS row padding per D#.
// dims/strides in elements; pad_interval: log2(dwords/row)-1; pad_amount: pad dwords-1.
__device__ __forceinline__ void tdm_load_2d(
    unsigned int lds_addr, unsigned long long gaddr,
    unsigned int tensor_d0, unsigned int tensor_d1,
    unsigned int tile_d0, unsigned int tile_d1,
    unsigned long long d0_stride,
    unsigned int pad_interval, unsigned int pad_amount)
{
    u32x4 g0;
    g0.x = 0x1u;                                            // count=1, user D#
    g0.y = lds_addr;                                        // LDS byte address
    g0.z = (unsigned int)(gaddr & 0xFFFFFFFFull);           // global_addr[31:0]
    g0.w = (unsigned int)((gaddr >> 32) & 0x01FFFFFFull) | (2u << 30); // [56:32] | type=2
    i32x8 g1;
    g1.s0 = (int)((1u << 16) | (1u << 20) |                 // data_size=2B, pad_enable
                  (pad_interval << 22) | (pad_amount << 25));
    g1.s1 = (int)((tensor_d0 & 0xFFFFu) << 16);             // tensor_dim0[15:0]
    g1.s2 = (int)(((tensor_d0 >> 16) & 0xFFFFu) | ((tensor_d1 & 0xFFFFu) << 16));
    g1.s3 = (int)(((tensor_d1 >> 16) & 0xFFFFu) | ((tile_d0 & 0xFFFFu) << 16));
    g1.s4 = (int)(tile_d1 & 0xFFFFu);                       // tile_dim1; tile_dim2=0
    g1.s5 = (int)(d0_stride & 0xFFFFFFFFull);               // dim0_stride[31:0]
    g1.s6 = (int)((d0_stride >> 32) & 0xFFFFull);           // dim0_stride[47:32]
    g1.s7 = 0;                                              // dim1_stride unused (2D)
    i32x4 z4 = {0, 0, 0, 0};
#if __clang_major__ >= 23
    i32x8 z8 = {0, 0, 0, 0, 0, 0, 0, 0};
    __builtin_amdgcn_tensor_load_to_lds(g0, g1, z4, z4, z8, 0);
#else
    __builtin_amdgcn_tensor_load_to_lds(g0, g1, z4, z4, 0);
#endif
}
#endif

// ---------------- prepass kernels: f32 -> bf16 workspace ----------------

// K: (b,s,h,e) f32 -> same layout bf16
__global__ __launch_bounds__(256) void prep_k_kernel(
    const float* __restrict__ K, unsigned short* __restrict__ out)
{
    const size_t idx = ((size_t)blockIdx.x * 256 + threadIdx.x) * 8;
    const float4 a = *(const float4*)(K + idx);
    const float4 c = *(const float4*)(K + idx + 4);
    uint4 u;
    u.x = cvt2(a.x, a.y); u.y = cvt2(a.z, a.w);
    u.z = cvt2(c.x, c.y); u.w = cvt2(c.z, c.w);
    *(uint4*)(out + idx) = u;
}

// V: (b,s,h,e) f32 -> transposed (b,h,e,s) bf16
__global__ __launch_bounds__(256) void prep_v_kernel(
    const float* __restrict__ V, unsigned short* __restrict__ out)
{
    __shared__ float T[64][65];
    const int bh = blockIdx.x >> 5;      // b*8+h
    const int s0 = (blockIdx.x & 31) * 64;
    const int b = bh >> 3, h = bh & 7;
    const int tid = threadIdx.x;
    #pragma unroll
    for (int i = 0; i < 4; ++i) {        // load 64s x 64e, coalesced in e
        const int c = tid + i * 256;
        const int row = c >> 4, col = (c & 15) << 2;
        const float4 v = *(const float4*)(V + ((((size_t)b * S_) + (s0 + row)) * H_ + h) * E_ + col);
        T[col + 0][row] = v.x; T[col + 1][row] = v.y;
        T[col + 2][row] = v.z; T[col + 3][row] = v.w;
    }
    __syncthreads();
    #pragma unroll
    for (int i = 0; i < 2; ++i) {        // write 64e x 64s, coalesced in s
        const int c = tid + i * 256;
        const int e = c >> 3, sc = (c & 7) << 3;
        uint4 u;
        u.x = cvt2(T[e][sc + 0], T[e][sc + 1]);
        u.y = cvt2(T[e][sc + 2], T[e][sc + 3]);
        u.z = cvt2(T[e][sc + 4], T[e][sc + 5]);
        u.w = cvt2(T[e][sc + 6], T[e][sc + 7]);
        *(uint4*)(out + ((size_t)(b * 8 + h) * 64 + e) * (size_t)S_ + s0 + sc) = u;
    }
}

// ---------------- main kernel: workspace bf16 + TDM staging ----------------

__global__ __launch_bounds__(128) void assa_attn_ws_kernel(
    const float* __restrict__ Q, const unsigned short* __restrict__ wsK,
    const unsigned short* __restrict__ wsV, const float* __restrict__ pa1,
    const float* __restrict__ pa2, float* __restrict__ O)
{
    __shared__ __attribute__((aligned(16))) unsigned short Kt[2][32 * KSTR];
    __shared__ __attribute__((aligned(16))) unsigned short Vt[2][64 * VSTR];  // V^T [e][s]
    __shared__ __attribute__((aligned(16))) unsigned int   PB[4][32 * PSTRU]; // lo=sparse hi=dense

    const int tid  = threadIdx.x;
    const int wave = tid >> 5;
    const int lane = tid & 31;
    const int half = lane >> 4;
    const int ln   = lane & 15;

    const int qTile = blockIdx.x & 31;
    const int bh    = blockIdx.x >> 5;
    const int h     = bh & 7;
    const int b     = bh >> 3;
    const int qBase = qTile * 64;

    const float w1 = __expf(pa1[0]);
    const float w2 = __expf(pa2[0]);
    const float alpha1 = w1 / (w1 + w2);
    const float alpha2 = w2 / (w1 + w2);
    const float c1 = alpha1 * LN2SQ;

    // per-(b,h) bf16 workspace bases (element offsets)
    const unsigned short* wsKb = wsK + (size_t)b * (S_ * H_ * E_) + (size_t)h * E_; // + s*512
    const unsigned short* wsVb = wsV + (size_t)(b * 8 + h) * 64 * (size_t)S_;       // + e*2048 + s

    // ---- Q A-fragments, with (1/sqrt(E)) * log2(e) folded in ----
    FragAB aQ0, aQ1;
    {
        const float scale = 0.125f * LOG2E;
        const int q = qBase + wave * 16 + ln;
        const float* qrow = Q + ((((size_t)b * L_) + q) * H_ + h) * E_;
        #pragma unroll
        for (int p = 0; p < 8; ++p) {
            const int k = a_pair_k(p, half);
            aQ0.u[p] = cvt2(qrow[k] * scale,      qrow[k + 1] * scale);
            aQ1.u[p] = cvt2(qrow[k + 32] * scale, qrow[k + 33] * scale);
        }
    }

    FragAB bOne;
    #pragma unroll
    for (int p = 0; p < 8; ++p) bOne.u[p] = 0x3F803F80u;

    FragC Os[4], Od[4], Ld;
    #pragma unroll
    for (int t = 0; t < 4; ++t)
        #pragma unroll
        for (int r = 0; r < 8; ++r) { Os[t].f[r] = 0.f; Od[t].f[r] = 0.f; }
    #pragma unroll
    for (int r = 0; r < 8; ++r) Ld.f[r] = 0.f;

    // ---- tile staging: TDM DMA (or synchronous bf16 copy fallback) ----
    auto stage_issue = [&](int sBase, int bsel) {
#if HAVE_TDM
        if (wave == 0) {
            // K tile: 32 rows (s) x 64 elems, row stride 512 elems; LDS pitch 144B
            tdm_load_2d((unsigned int)(size_t)&Kt[bsel][0],
                        (unsigned long long)(size_t)(wsKb + (size_t)sBase * (H_ * E_)),
                        E_, S_, E_, 32, H_ * E_,
                        /*pad_interval=*/4, /*pad_amount=*/3); // 32dw rows + 4dw pad
            // V^T tile: 64 rows (e) x 32 elems (s), row stride 2048 elems; LDS pitch 80B
            tdm_load_2d((unsigned int)(size_t)&Vt[bsel][0],
                        (unsigned long long)(size_t)(wsVb + sBase),
                        S_, E_, 32, E_, S_,
                        /*pad_interval=*/3, /*pad_amount=*/3); // 16dw rows + 4dw pad
        }
#else
        // synchronous cooperative copy of bf16 tiles
        #pragma unroll
        for (int i = 0; i < 2; ++i) {    // K: 32 rows x 8 chunks of 16B
            const int c = tid + i * 128;
            const int row = c >> 3, off = (c & 7) << 3;
            *(uint4*)&Kt[bsel][row * KSTR + off] =
                *(const uint4*)(wsKb + (size_t)(sBase + row) * (H_ * E_) + off);
        }
        #pragma unroll
        for (int i = 0; i < 2; ++i) {    // V^T: 64 rows x 4 chunks of 16B
            const int c = tid + i * 128;
            const int row = c >> 2, off = (c & 3) << 3;
            *(uint4*)&Vt[bsel][row * VSTR + off] =
                *(const uint4*)(wsVb + (size_t)row * S_ + sBase + off);
        }
#endif
    };
    auto stage_fence = [&]() {
#if HAVE_TDM
        if (wave == 0) __builtin_amdgcn_s_wait_tensorcnt(0);
#endif
    };

    // prologue: tile 0
    stage_issue(0, 0);
    stage_fence();
    __syncthreads();

    int cur = 0;
    for (int it = 0; it < NIT; ++it) {
        const bool has_next = (it + 1) < NIT;
        if (has_next) {
            stage_issue((it + 1) * 32, cur ^ 1);   // DMA next tile during compute
            if (it + 2 < NIT) {                    // warm L2 two tiles ahead
                __builtin_prefetch(wsKb + (size_t)((it + 2) * 32 + (tid & 31)) * (H_ * E_), 0, 1);
                __builtin_prefetch(wsVb + (size_t)(tid & 63) * S_ + (it + 2) * 32, 0, 1);
            }
        }

        // ---- QK^T: batch all four B fragments, then 4 WMMAs ----
        FragAB bK[4];
        #pragma unroll
        for (int jt = 0; jt < 2; ++jt) {
            const int rowk = (jt * 16 + ln) * KSTR;
            #pragma unroll
            for (int p = 0; p < 8; ++p) {
                bK[jt * 2 + 0].u[p] = *(const unsigned int*)&Kt[cur][rowk +      half * 16 + 2 * p];
                bK[jt * 2 + 1].u[p] = *(const unsigned int*)&Kt[cur][rowk + 32 + half * 16 + 2 * p];
            }
        }
        FragC c01[2];
        #pragma unroll
        for (int jt = 0; jt < 2; ++jt) {
            FragC c;
            #pragma unroll
            for (int r = 0; r < 8; ++r) c.f[r] = 0.f;
            c.v = __builtin_amdgcn_wmma_f32_16x16x32_bf16(
                      false, aQ0.v, false, bK[jt * 2 + 0].v, (short)0, c.v, false, false);
            c.v = __builtin_amdgcn_wmma_f32_16x16x32_bf16(
                      false, aQ1.v, false, bK[jt * 2 + 1].v, (short)0, c.v, false, false);
            c01[jt] = c;
        }

        // ---- scores: pe = exp2(s'), pr = c1*relu(s')^2 (relu via v_med3) ----
        #pragma unroll
        for (int jt = 0; jt < 2; ++jt) {
            unsigned int pk8[8];
            #pragma unroll
            for (int r = 0; r < 8; ++r) {
                const float s  = c01[jt].f[r];
                const float pe = __builtin_amdgcn_exp2f(s);
                const float t  = __builtin_amdgcn_fmed3f(s, 0.f, __builtin_inff());
                const float pr = (c1 * t) * t;
                pk8[r] = cvt2(pr, pe);
            }
            unsigned int* dst = &PB[wave][(jt * 16 + ln) * PSTRU + half * 8];
            *(uint4*)(dst)     = make_uint4(pk8[0], pk8[1], pk8[2], pk8[3]);
            *(uint4*)(dst + 4) = make_uint4(pk8[4], pk8[5], pk8[6], pk8[7]);
        }

        // ---- P column-major -> A-layout; one b32 pair feeds both branches ----
        FragAB aPs, aPd;
        #pragma unroll
        for (int p = 0; p < 8; ++p) {
            const int k = a_pair_k(p, half);
            const unsigned int u0 = PB[wave][k * PSTRU + ln];
            const unsigned int u1 = PB[wave][(k + 1) * PSTRU + ln];
            aPs.u[p] = __builtin_amdgcn_perm(u1, u0, 0x05040100u);
            aPd.u[p] = __builtin_amdgcn_perm(u1, u0, 0x07060302u);
        }

        // softmax row sums on the matrix pipe
        Ld.v = __builtin_amdgcn_wmma_f32_16x16x32_bf16(
                   false, aPd.v, false, bOne.v, (short)0, Ld.v, false, false);

        // ---- P @ V : 4 output N-tiles x 2 branches ----
        #pragma unroll
        for (int tp = 0; tp < 2; ++tp) {
            FragAB bV[2];
            #pragma unroll
            for (int j = 0; j < 2; ++j) {
                const int t = tp * 2 + j;
                #pragma unroll
                for (int p = 0; p < 8; ++p)
                    bV[j].u[p] = *(const unsigned int*)&Vt[cur][(t * 16 + ln) * VSTR + half * 16 + 2 * p];
            }
            #pragma unroll
            for (int j = 0; j < 2; ++j) {
                const int t = tp * 2 + j;
                Os[t].v = __builtin_amdgcn_wmma_f32_16x16x32_bf16(
                              false, aPs.v, false, bV[j].v, (short)0, Os[t].v, false, false);
                Od[t].v = __builtin_amdgcn_wmma_f32_16x16x32_bf16(
                              false, aPd.v, false, bV[j].v, (short)0, Od[t].v, false, false);
            }
        }

        stage_fence();      // TENSORcnt drain before making next tile visible
        __syncthreads();
        cur ^= 1;
    }

    // ---- epilogue: Ld holds row sums replicated across lanes ----
    #pragma unroll
    for (int r = 0; r < 8; ++r) {
        const float wgt = alpha2 * __builtin_amdgcn_rcpf(Ld.f[r]);
        const int row = r + half * 8;
        const int q   = qBase + wave * 16 + row;
        float* orow = O + ((((size_t)b * L_) + q) * H_ + h) * E_;
        #pragma unroll
        for (int t = 0; t < 4; ++t)
            orow[t * 16 + ln] = fmaf(wgt, Od[t].f[r], Os[t].f[r]);
    }
}

// ---------------- legacy kernel (no-workspace fallback) ----------------

__global__ __launch_bounds__(128) void assa_attn_legacy_kernel(
    const float* __restrict__ Q, const float* __restrict__ K,
    const float* __restrict__ V, const float* __restrict__ pa1,
    const float* __restrict__ pa2, float* __restrict__ O)
{
    __shared__ __attribute__((aligned(16))) unsigned short Kt[2][32 * KSTR];
    __shared__ __attribute__((aligned(16))) unsigned short Vt[2][64 * VSTR];
    __shared__ __attribute__((aligned(16))) unsigned int   PB[4][32 * PSTRU];

    const int tid  = threadIdx.x;
    const int wave = tid >> 5;
    const int lane = tid & 31;
    const int half = lane >> 4;
    const int ln   = lane & 15;

    const int qTile = blockIdx.x & 31;
    const int bh    = blockIdx.x >> 5;
    const int h     = bh & 7;
    const int b     = bh >> 3;
    const int qBase = qTile * 64;

    const float w1 = __expf(pa1[0]);
    const float w2 = __expf(pa2[0]);
    const float alpha1 = w1 / (w1 + w2);
    const float alpha2 = w2 / (w1 + w2);
    const float c1 = alpha1 * LN2SQ;

    FragAB aQ0, aQ1;
    {
        const float scale = 0.125f * LOG2E;
        const int q = qBase + wave * 16 + ln;
        const float* qrow = Q + ((((size_t)b * L_) + q) * H_ + h) * E_;
        #pragma unroll
        for (int p = 0; p < 8; ++p) {
            const int k = a_pair_k(p, half);
            aQ0.u[p] = cvt2(qrow[k] * scale,      qrow[k + 1] * scale);
            aQ1.u[p] = cvt2(qrow[k + 32] * scale, qrow[k + 33] * scale);
        }
    }

    FragAB bOne;
    #pragma unroll
    for (int p = 0; p < 8; ++p) bOne.u[p] = 0x3F803F80u;

    FragC Os[4], Od[4], Ld;
    #pragma unroll
    for (int t = 0; t < 4; ++t)
        #pragma unroll
        for (int r = 0; r < 8; ++r) { Os[t].f[r] = 0.f; Od[t].f[r] = 0.f; }
    #pragma unroll
    for (int r = 0; r < 8; ++r) Ld.f[r] = 0.f;

    const float* Kb = K + (((size_t)b * S_) * H_ + h) * E_;
    const float* Vb = V + (((size_t)b * S_) * H_ + h) * E_;

    float4 kf[4], vf[4];

    auto load_tile = [&](int s0) {
        #pragma unroll
        for (int i = 0; i < 2; ++i) {
            const int c = tid + i * 128;
            const int row = c >> 3, col = (c & 7) << 3;
            const float* p = Kb + (size_t)(s0 + row) * (H_ * E_) + col;
            kf[2 * i]     = *(const float4*)p;
            kf[2 * i + 1] = *(const float4*)(p + 4);
        }
        #pragma unroll
        for (int i = 0; i < 4; ++i) {
            const int c = tid + i * 128;
            const int row = c >> 4, col = (c & 15) << 2;
            vf[i] = *(const float4*)(Vb + (size_t)(s0 + row) * (H_ * E_) + col);
        }
    };
    auto stage = [&](int bsel) {
        #pragma unroll
        for (int i = 0; i < 2; ++i) {
            const int c = tid + i * 128;
            const int row = c >> 3, col = (c & 7) << 3;
            uint4 u;
            u.x = cvt2(kf[2 * i].x,     kf[2 * i].y);
            u.y = cvt2(kf[2 * i].z,     kf[2 * i].w);
            u.z = cvt2(kf[2 * i + 1].x, kf[2 * i + 1].y);
            u.w = cvt2(kf[2 * i + 1].z, kf[2 * i + 1].w);
            *(uint4*)&Kt[bsel][row * KSTR + col] = u;
        }
        #pragma unroll
        for (int i = 0; i < 4; ++i) {
            const int c = tid + i * 128;
            const int row = c >> 4, col = (c & 15) << 2;
            const unsigned int p0 = cvt2(vf[i].x, vf[i].y);
            const unsigned int p1 = cvt2(vf[i].z, vf[i].w);
            Vt[bsel][(col + 0) * VSTR + row] = (unsigned short)p0;
            Vt[bsel][(col + 1) * VSTR + row] = (unsigned short)(p0 >> 16);
            Vt[bsel][(col + 2) * VSTR + row] = (unsigned short)p1;
            Vt[bsel][(col + 3) * VSTR + row] = (unsigned short)(p1 >> 16);
        }
    };

    load_tile(0);
    stage(0);
    __syncthreads();

    int cur = 0;
    for (int it = 0; it < NIT; ++it) {
        const bool has_next = (it + 1) < NIT;
        if (has_next) load_tile((it + 1) * 32);

        FragAB bK[4];
        #pragma unroll
        for (int jt = 0; jt < 2; ++jt) {
            const int rowk = (jt * 16 + ln) * KSTR;
            #pragma unroll
            for (int p = 0; p < 8; ++p) {
                bK[jt * 2 + 0].u[p] = *(const unsigned int*)&Kt[cur][rowk +      half * 16 + 2 * p];
                bK[jt * 2 + 1].u[p] = *(const unsigned int*)&Kt[cur][rowk + 32 + half * 16 + 2 * p];
            }
        }
        FragC c01[2];
        #pragma unroll
        for (int jt = 0; jt < 2; ++jt) {
            FragC c;
            #pragma unroll
            for (int r = 0; r < 8; ++r) c.f[r] = 0.f;
            c.v = __builtin_amdgcn_wmma_f32_16x16x32_bf16(
                      false, aQ0.v, false, bK[jt * 2 + 0].v, (short)0, c.v, false, false);
            c.v = __builtin_amdgcn_wmma_f32_16x16x32_bf16(
                      false, aQ1.v, false, bK[jt * 2 + 1].v, (short)0, c.v, false, false);
            c01[jt] = c;
        }

        #pragma unroll
        for (int jt = 0; jt < 2; ++jt) {
            unsigned int pk8[8];
            #pragma unroll
            for (int r = 0; r < 8; ++r) {
                const float s  = c01[jt].f[r];
                const float pe = __builtin_amdgcn_exp2f(s);
                const float t  = __builtin_amdgcn_fmed3f(s, 0.f, __builtin_inff());
                const float pr = (c1 * t) * t;
                pk8[r] = cvt2(pr, pe);
            }
            unsigned int* dst = &PB[wave][(jt * 16 + ln) * PSTRU + half * 8];
            *(uint4*)(dst)     = make_uint4(pk8[0], pk8[1], pk8[2], pk8[3]);
            *(uint4*)(dst + 4) = make_uint4(pk8[4], pk8[5], pk8[6], pk8[7]);
        }

        FragAB aPs, aPd;
        #pragma unroll
        for (int p = 0; p < 8; ++p) {
            const int k = a_pair_k(p, half);
            const unsigned int u0 = PB[wave][k * PSTRU + ln];
            const unsigned int u1 = PB[wave][(k + 1) * PSTRU + ln];
            aPs.u[p] = __builtin_amdgcn_perm(u1, u0, 0x05040100u);
            aPd.u[p] = __builtin_amdgcn_perm(u1, u0, 0x07060302u);
        }

        Ld.v = __builtin_amdgcn_wmma_f32_16x16x32_bf16(
                   false, aPd.v, false, bOne.v, (short)0, Ld.v, false, false);

        #pragma unroll
        for (int tp = 0; tp < 2; ++tp) {
            FragAB bV[2];
            #pragma unroll
            for (int j = 0; j < 2; ++j) {
                const int t = tp * 2 + j;
                #pragma unroll
                for (int p = 0; p < 8; ++p)
                    bV[j].u[p] = *(const unsigned int*)&Vt[cur][(t * 16 + ln) * VSTR + half * 16 + 2 * p];
            }
            #pragma unroll
            for (int j = 0; j < 2; ++j) {
                const int t = tp * 2 + j;
                Os[t].v = __builtin_amdgcn_wmma_f32_16x16x32_bf16(
                              false, aPs.v, false, bV[j].v, (short)0, Os[t].v, false, false);
                Od[t].v = __builtin_amdgcn_wmma_f32_16x16x32_bf16(
                              false, aPd.v, false, bV[j].v, (short)0, Od[t].v, false, false);
            }
        }

        if (has_next) stage(cur ^ 1);
        __syncthreads();
        cur ^= 1;
    }

    #pragma unroll
    for (int r = 0; r < 8; ++r) {
        const float wgt = alpha2 * __builtin_amdgcn_rcpf(Ld.f[r]);
        const int row = r + half * 8;
        const int q   = qBase + wave * 16 + row;
        float* orow = O + ((((size_t)b * L_) + q) * H_ + h) * E_;
        #pragma unroll
        for (int t = 0; t < 4; ++t)
            orow[t * 16 + ln] = fmaf(wgt, Od[t].f[r], Os[t].f[r]);
    }
}

extern "C" void kernel_launch(void* const* d_in, const int* in_sizes, int n_in,
                              void* d_out, int out_size, void* d_ws, size_t ws_size,
                              hipStream_t stream) {
    const float* Q  = (const float*)d_in[0];
    const float* K  = (const float*)d_in[1];
    const float* V  = (const float*)d_in[2];
    const float* a1 = (const float*)d_in[3];
    const float* a2 = (const float*)d_in[4];
    float* O = (float*)d_out;

    const size_t kvElems = (size_t)B_ * S_ * H_ * E_;      // 2,097,152
    const size_t need    = kvElems * 2 /*bf16*/ * 2 /*K+V*/;

    dim3 grid(B_ * H_ * (L_ / 64)), block(128);

    if (d_ws != nullptr && ws_size >= need) {
        unsigned short* wsK = (unsigned short*)d_ws;
        unsigned short* wsV = wsK + kvElems;
        hipLaunchKernelGGL(prep_k_kernel, dim3(kvElems / (256 * 8)), dim3(256), 0, stream, K, wsK);
        hipLaunchKernelGGL(prep_v_kernel, dim3(B_ * H_ * (S_ / 64)), dim3(256), 0, stream, V, wsV);
        hipLaunchKernelGGL(assa_attn_ws_kernel, grid, block, 0, stream, Q, wsK, wsV, a1, a2, O);
    } else {
        hipLaunchKernelGGL(assa_attn_legacy_kernel, grid, block, 0, stream, Q, K, V, a1, a2, O);
    }
}
